// TreeTransformer_16458314678994
// MI455X (gfx1250) — matrix operand order repaired and checked by reading, not compiled, not run on hardware
//
#include <hip/hip_runtime.h>
#include <hip/hip_bf16.h>
#include <stdint.h>

typedef _Float16 h16;
typedef _Float16 v16h __attribute__((ext_vector_type(16)));
typedef float    v8f  __attribute__((ext_vector_type(8)));
typedef int      gv4i __attribute__((vector_size(16)));  // matches builtin param type

// CDNA5 async global->LDS path (ASYNCcnt-tracked), device pass only.
#if defined(__HIP_DEVICE_COMPILE__) && \
    __has_builtin(__builtin_amdgcn_global_load_async_to_lds_b128) && \
    __has_builtin(__builtin_amdgcn_s_wait_asynccnt)
#define GFXASYNC 1
#else
#define GFXASYNC 0
#endif
#define AS1 __attribute__((address_space(1)))
#define AS3 __attribute__((address_space(3)))

namespace {
constexpr int Bb = 64, Cc = 512, Tt = 256, Hh = 8, Dd = 64, FFNn = 1024, OUTC = 256, NL = 4;
constexpr int MT = Bb * Tt;           // 16384 rows of the residual stream
constexpr int NNODES = Tt - 1;        // 255
}

// ---------------------------------------------------------------------------
// Generic batched WMMA GEMM:  D[m][n] = epi( sum_k A[m][k] * B[n][k] )
// A: row stride sa, batch offset (z/ZH)*aS1 + (z%ZH)*aS2   (f16 or f32)
// B: row stride sb (contiguous in k), batched likewise     (f16)
// Block: 256 threads = 8 wave32; macro tile 128x64, K step 32, double-buffered
// LDS (async-to-LDS when available). Wave w: rows [16w,16w+16) x 64 cols.
// EPI: 0 bias->f16 | 1 bias+gelu->f16 | 2 bias+res->f32 | 3 scale+mask->f16
//      4 plain->f16 | 5 bias->f32
// ---------------------------------------------------------------------------
struct GArgs {
  const void* A;  long aS1, aS2; int sa;
  const h16*  Bm; long bS1, bS2; int sb;
  void*       O;  long oS1, oS2; int so;
  const float* bias;
  const float* res;                       // same indexing as O (EPI 2)
  const float* mask; long mS1, mS2;       // ld = N (EPI 3)
  int M, N, K, ZH, NZ;
  float scale;
};

template <int EPI, bool AF32>
__global__ __launch_bounds__(256) void gemm_wmma(GArgs g) {
  __shared__ h16 lA[2][128 * 32];  // [buf][row][k]
  __shared__ h16 lB[2][64 * 32];   // [buf][n][k]

  const int tid  = threadIdx.x;
  const int wave = tid >> 5;
  const int lane = tid & 31;
  const bool hi  = lane >= 16;
  const int l16  = lane & 15;

  const int z  = blockIdx.z;
  const int zq = z / g.ZH, zr = z % g.ZH;
  const long aOff = (long)zq * g.aS1 + (long)zr * g.aS2;
  const long bOff = (long)zq * g.bS1 + (long)zr * g.bS2;
  const long oOff = (long)zq * g.oS1 + (long)zr * g.oS2;
  const int mBase = blockIdx.y * 128;
  const int nBase = blockIdx.x * 64;

  const int nK = g.K >> 5;
  // Per-thread staging slots: A tile 128 rows x 2 segs of 16 halves (32B),
  //                           B tile 64 rows x 4 segs of 8 halves (16B).
  const int arow = tid >> 1, aseg = tid & 1;
  const int brow = tid >> 2, bseg = tid & 3;
  const int gRowA = mBase + arow;
  const bool aOK = gRowA < g.M;
  const h16*   aP = (const h16*)g.A + aOff + (long)gRowA * g.sa + aseg * 16;
  const float* aF = (const float*)g.A + aOff + (long)gRowA * g.sa + aseg * 16;
  const h16*   bP = g.Bm + bOff + (long)(nBase + brow) * g.sb + bseg * 8;
  const int aSlot = arow * 32 + aseg * 16;
  const int bSlot = brow * 32 + bseg * 8;

  v8f acc[4] = {};

  // ---- fragment build + 4 back-to-back WMMAs on one LDS buffer ----
  auto doTile = [&](int buf) {
    struct AF8 { unsigned u[8]; } af;
    const int mloc = wave * 16 + l16;
#pragma unroll
    for (int v = 0; v < 8; ++v) {  // lane holds K=(hi?16:0)+j for row m, 8 x ds_load_b32
      const int kk = (hi ? 16 : 0) + v * 2;
      af.u[v] = *(const unsigned*)&lA[buf][mloc * 32 + kk];
    }
    const v16h av = __builtin_bit_cast(v16h, af);
    v16h bv[4];
#pragma unroll
    for (int nt = 0; nt < 4; ++nt) {  // lane holds K=(hi?16:0)+j for col n, 2 x ds_load_b128
      struct BF { uint4 a, b; } bf;
      const h16* pb = &lB[buf][(nt * 16 + l16) * 32 + (hi ? 16 : 0)];
      bf.a = *(const uint4*)pb;
      bf.b = *(const uint4*)(pb + 8);
      bv[nt] = __builtin_bit_cast(v16h, bf);
    }
#pragma unroll
    for (int nt = 0; nt < 4; ++nt)
      acc[nt] = __builtin_amdgcn_wmma_f32_16x16x32_f16(false, av, false, bv[nt],
                                                       (short)0, acc[nt], false, false);
  };

#if GFXASYNC
  if constexpr (!AF32) {
    // ---- async global->LDS double-buffered pipeline ----
    auto issue = [&](int kt, int buf) {
      const int k0 = kt << 5;
      if (aOK) {
        __builtin_amdgcn_global_load_async_to_lds_b128(
            (AS1 gv4i*)(aP + k0), (AS3 gv4i*)(&lA[buf][aSlot]), 0, 0);
        __builtin_amdgcn_global_load_async_to_lds_b128(
            (AS1 gv4i*)(aP + k0 + 8), (AS3 gv4i*)(&lA[buf][aSlot + 8]), 0, 0);
      } else {
        uint4 zz = make_uint4(0u, 0u, 0u, 0u);
        *(uint4*)&lA[buf][aSlot] = zz;
        *(uint4*)&lA[buf][aSlot + 8] = zz;
      }
      __builtin_amdgcn_global_load_async_to_lds_b128(
          (AS1 gv4i*)(bP + k0), (AS3 gv4i*)(&lB[buf][bSlot]), 0, 0);
    };
    issue(0, 0);
    for (int kt = 0; kt < nK; ++kt) {
      const int cur = kt & 1;
      __builtin_amdgcn_s_wait_asynccnt(0);  // my fills of buf cur are done
      __syncthreads();                      // everyone's fills done; prev readers of nxt done
      if (kt + 1 < nK) issue(kt + 1, cur ^ 1);
      doTile(cur);
    }
  } else
#endif
  {
    // ---- register-staged double-buffered pipeline (fallback / f32-A path) ----
    uint4 ra0 = {}, ra1 = {}, rb = {};
    float4 f0 = {}, f1 = {}, f2 = {}, f3 = {};
    auto loadRegs = [&](int kt) {
      const int k0 = kt << 5;
      if constexpr (AF32) {
        if (aOK) {
          const float* ap = aF + k0;
          f0 = *(const float4*)ap;
          f1 = *(const float4*)(ap + 4);
          f2 = *(const float4*)(ap + 8);
          f3 = *(const float4*)(ap + 12);
        }
      } else {
        if (aOK) {
          ra0 = *(const uint4*)(aP + k0);
          ra1 = *(const uint4*)(aP + k0 + 8);
          if (k0 + 64 < g.K) __builtin_prefetch(aP + k0 + 32, 0, 0);  // global_prefetch_b8
        }
      }
      rb = *(const uint4*)(bP + k0);
    };
    auto storeRegs = [&](int buf) {
      h16* dst = &lA[buf][aSlot];
      if (aOK) {
        if constexpr (AF32) {
          dst[0]  = (h16)f0.x; dst[1]  = (h16)f0.y; dst[2]  = (h16)f0.z; dst[3]  = (h16)f0.w;
          dst[4]  = (h16)f1.x; dst[5]  = (h16)f1.y; dst[6]  = (h16)f1.z; dst[7]  = (h16)f1.w;
          dst[8]  = (h16)f2.x; dst[9]  = (h16)f2.y; dst[10] = (h16)f2.z; dst[11] = (h16)f2.w;
          dst[12] = (h16)f3.x; dst[13] = (h16)f3.y; dst[14] = (h16)f3.z; dst[15] = (h16)f3.w;
        } else {
          *(uint4*)dst       = ra0;
          *(uint4*)(dst + 8) = ra1;
        }
      } else {
        uint4 zz = make_uint4(0u, 0u, 0u, 0u);
        *(uint4*)dst       = zz;
        *(uint4*)(dst + 8) = zz;
      }
      *(uint4*)&lB[buf][bSlot] = rb;
    };
    loadRegs(0);
    storeRegs(0);
    for (int kt = 0; kt < nK; ++kt) {
      const int cur = kt & 1;
      if (kt + 1 < nK) loadRegs(kt + 1);  // overlap global latency with compute
      __syncthreads();
      doTile(cur);
      if (kt + 1 < nK) storeRegs(cur ^ 1);
    }
  }

  // ---- epilogue (C/D layout: VGPR r -> M = r + (hi?8:0), N = l16) ----
#pragma unroll
  for (int nt = 0; nt < 4; ++nt) {
#pragma unroll
    for (int r = 0; r < 8; ++r) {
      const int gRow = mBase + wave * 16 + r + (hi ? 8 : 0);
      if (gRow >= g.M) continue;
      const int gCol = nBase + nt * 16 + l16;
      float v = acc[nt][r];
      const long oi = oOff + (long)gRow * g.so + gCol;
      if (EPI == 0) {
        v += g.bias[gCol];
        ((h16*)g.O)[oi] = (h16)v;
      } else if (EPI == 1) {
        v += g.bias[gCol];
        v = 0.5f * v * (1.0f + erff(v * 0.70710678118654752f));  // exact GELU
        ((h16*)g.O)[oi] = (h16)v;
      } else if (EPI == 2) {
        v += g.bias[gCol] + g.res[oi];
        ((float*)g.O)[oi] = v;
      } else if (EPI == 3) {
        const long mi = (long)zq * g.mS1 + (long)zr * g.mS2 + (long)gRow * g.N + gCol;
        v = v * g.scale + g.mask[mi];
        ((h16*)g.O)[oi] = (h16)v;
      } else if (EPI == 4) {
        ((h16*)g.O)[oi] = (h16)v;
      } else {
        v += g.bias[gCol];
        ((float*)g.O)[oi] = v;
      }
    }
  }
}

// ---------------------------------------------------------------------------
// Small helper kernels
// ---------------------------------------------------------------------------
__global__ void mask_init_k(float* mask, long total) {  // (B,T,T): -100, diag 0
  const long i = (long)blockIdx.x * 256 + threadIdx.x;
  if (i >= total) return;
  const int kcol = (int)(i & 255);
  const int qrow = (int)((i >> 8) & 255);
  mask[i] = (kcol == qrow) ? 0.0f : -100.0f;
}

__global__ void mask_scatter_k(const long long* idx, float* mask) {
  const int i = blockIdx.x * 256 + threadIdx.x;
  if (i >= Bb * NNODES) return;
  const int b = i / NNODES, node = i % NNODES;
  const int pos = node + 1;
  const long mb = (long)b * Tt * Tt;
#pragma unroll
  for (int j = 0; j < 3; ++j) {
    const long long t = idx[(long)b * 3 * NNODES + (long)node * 3 + j];
    const int rel = (t > 0) ? (int)t : pos;
    mask[mb + (long)pos * Tt + rel] = 0.0f;  // idempotent 0-stores: races benign
    mask[mb + (long)rel * Tt + pos] = 0.0f;
  }
}

__global__ void transpose_trees_k(const float* __restrict__ trees, float* __restrict__ x) {
  const long i = (long)blockIdx.x * 256 + threadIdx.x;  // over B*T*C
  if (i >= (long)Bb * Tt * Cc) return;
  const int c = (int)(i % Cc);
  const long r = i / Cc;
  const int t = (int)(r % Tt);
  const long b = r / Tt;
  x[i] = trees[(b * Cc + c) * Tt + t];
}

__global__ void convw_k(const float* __restrict__ W, h16* __restrict__ Wt, int K, int N) {
  const long i = (long)blockIdx.x * 256 + threadIdx.x;  // W is (K,N) -> Wt (N,K)
  if (i >= (long)K * N) return;
  const int n = (int)(i % N);
  const long k = i / N;
  Wt[(long)n * K + k] = (h16)W[i];
}

__global__ void transv_k(const h16* __restrict__ v16, h16* __restrict__ Vt) {
  const long i = (long)blockIdx.x * 256 + threadIdx.x;  // over B*T*C
  if (i >= (long)Bb * Tt * Cc) return;
  const int c = (int)(i % Cc);
  const long r = i / Cc;
  const int t = (int)(r % Tt);
  const long b = r / Tt;
  const int h = c >> 6, d = c & 63;
  Vt[(((b * Hh + h) * Dd + d) << 8) + t] = v16[i];  // (B,H,D,T)
}

__global__ __launch_bounds__(256) void ln_k(const float* __restrict__ x,
                                            const float* __restrict__ gw,
                                            const float* __restrict__ bw,
                                            h16* __restrict__ y) {
  __shared__ float red[256];
  const long base = (long)blockIdx.x * Cc;
  const int t = threadIdx.x;
  const float a = x[base + t], b2 = x[base + t + 256];
  red[t] = a + b2;
  __syncthreads();
  for (int s = 128; s > 0; s >>= 1) { if (t < s) red[t] += red[t + s]; __syncthreads(); }
  const float mu = red[0] * (1.0f / Cc);
  __syncthreads();
  const float da = a - mu, db = b2 - mu;
  red[t] = da * da + db * db;
  __syncthreads();
  for (int s = 128; s > 0; s >>= 1) { if (t < s) red[t] += red[t + s]; __syncthreads(); }
  const float rstd = rsqrtf(red[0] * (1.0f / Cc) + 1e-5f);
  y[base + t]       = (h16)(da * rstd * gw[t] + bw[t]);
  y[base + t + 256] = (h16)(db * rstd * gw[t + 256] + bw[t + 256]);
}

__global__ __launch_bounds__(256) void softmax_k(h16* S) {  // in-place, rows of 256
  __shared__ float red[256];
  const long base = (long)blockIdx.x * Tt;
  const int t = threadIdx.x;
  const float v = (float)S[base + t];
  red[t] = v;
  __syncthreads();
  for (int s = 128; s > 0; s >>= 1) { if (t < s) red[t] = fmaxf(red[t], red[t + s]); __syncthreads(); }
  const float mx = red[0];
  __syncthreads();
  const float e = __expf(v - mx);
  red[t] = e;
  __syncthreads();
  for (int s = 128; s > 0; s >>= 1) { if (t < s) red[t] += red[t + s]; __syncthreads(); }
  S[base + t] = (h16)(e / red[0]);
}

// ---------------------------------------------------------------------------
template <int EPI, bool AF32>
static void rg(const GArgs& g, hipStream_t s) {
  dim3 grid((unsigned)(g.N / 64), (unsigned)((g.M + 127) / 128), (unsigned)g.NZ);
  gemm_wmma<EPI, AF32><<<grid, dim3(256), 0, s>>>(g);
}

extern "C" void kernel_launch(void* const* d_in, const int* in_sizes, int n_in,
                              void* d_out, int out_size, void* d_ws, size_t ws_size,
                              hipStream_t stream) {
  (void)in_sizes; (void)n_in; (void)out_size; (void)ws_size;
  const float* trees = (const float*)d_in[0];
  const long long* idxes = (const long long*)d_in[1];
  // params in insertion order: per layer {ln1_g,ln1_b,Wq,bq,Wk,bk,Wv,bv,Wo,bo,ln2_g,ln2_b,W1,b1,W2,b2}
  auto LP = [&](int l, int j) -> const float* { return (const float*)d_in[2 + l * 16 + j]; };
  const float* Wout = (const float*)d_in[2 + NL * 16 + 0];
  const float* bout = (const float*)d_in[2 + NL * 16 + 1];

  // ---- workspace carve ----
  char* wp = (char*)d_ws;
  auto carve = [&](size_t n) { char* r = wp; wp += (n + 255) & ~(size_t)255; return r; };
  float* mask = (float*)carve(sizeof(float) * (size_t)Bb * Tt * Tt);
  float* x    = (float*)carve(sizeof(float) * (size_t)MT * Cc);
  h16* y16    = (h16*)carve(sizeof(h16) * (size_t)MT * Cc);
  h16* q16    = (h16*)carve(sizeof(h16) * (size_t)MT * Cc);
  h16* k16    = (h16*)carve(sizeof(h16) * (size_t)MT * Cc);   // contiguous after q16
  h16* v16p   = (h16*)carve(sizeof(h16) * (size_t)MT * Cc);
  h16* Vt     = (h16*)carve(sizeof(h16) * (size_t)MT * Cc);
  h16* S      = (h16*)carve(sizeof(h16) * (size_t)Bb * Hh * Tt * Tt);
  h16* wqT[NL]; h16* wkT[NL]; h16* wvT[NL]; h16* woT[NL]; h16* w1T[NL]; h16* w2T[NL];
  for (int l = 0; l < NL; ++l) {
    wqT[l] = (h16*)carve(sizeof(h16) * Cc * Cc);
    wkT[l] = (h16*)carve(sizeof(h16) * Cc * Cc);
    wvT[l] = (h16*)carve(sizeof(h16) * Cc * Cc);
    woT[l] = (h16*)carve(sizeof(h16) * Cc * Cc);
    w1T[l] = (h16*)carve(sizeof(h16) * Cc * FFNn);
    w2T[l] = (h16*)carve(sizeof(h16) * FFNn * Cc);
  }
  h16* woutT = (h16*)carve(sizeof(h16) * Cc * OUTC);
  h16* o16  = y16;  // alias: y16 dead between QKV and LN2
  h16* hffn = q16;  // alias: spans q16+k16 (dead after score GEMM)

  // ---- one-time prep ----
  {
    const long mt = (long)Bb * Tt * Tt;
    mask_init_k<<<(unsigned)((mt + 255) / 256), 256, 0, stream>>>(mask, mt);
    mask_scatter_k<<<(Bb * NNODES + 255) / 256, 256, 0, stream>>>(idxes, mask);
    transpose_trees_k<<<(MT * Cc) / 256, 256, 0, stream>>>(trees, x);
    for (int l = 0; l < NL; ++l) {
      convw_k<<<(Cc * Cc) / 256, 256, 0, stream>>>(LP(l, 2), wqT[l], Cc, Cc);
      convw_k<<<(Cc * Cc) / 256, 256, 0, stream>>>(LP(l, 4), wkT[l], Cc, Cc);
      convw_k<<<(Cc * Cc) / 256, 256, 0, stream>>>(LP(l, 6), wvT[l], Cc, Cc);
      convw_k<<<(Cc * Cc) / 256, 256, 0, stream>>>(LP(l, 8), woT[l], Cc, Cc);
      convw_k<<<(Cc * FFNn) / 256, 256, 0, stream>>>(LP(l, 12), w1T[l], Cc, FFNn);
      convw_k<<<(FFNn * Cc) / 256, 256, 0, stream>>>(LP(l, 14), w2T[l], FFNn, Cc);
    }
    convw_k<<<(Cc * OUTC) / 256, 256, 0, stream>>>(Wout, woutT, Cc, OUTC);
  }

  // ---- encoder layers ----
  for (int l = 0; l < NL; ++l) {
    ln_k<<<MT, 256, 0, stream>>>(x, LP(l, 0), LP(l, 1), y16);

    GArgs g{};
    g.A = y16; g.sa = Cc; g.ZH = 1; g.NZ = 1;
    g.M = MT; g.N = Cc; g.K = Cc; g.so = Cc;
    g.Bm = wqT[l]; g.sb = Cc; g.O = q16; g.bias = LP(l, 3); rg<0, false>(g, stream);
    g.Bm = wkT[l];             g.O = k16; g.bias = LP(l, 5); rg<0, false>(g, stream);
    g.Bm = wvT[l];             g.O = v16p; g.bias = LP(l, 7); rg<0, false>(g, stream);

    {  // scores: S(b,h) = (Q Kt) * d^-1/2 + mask(b)   -> f16, in-place softmax later
      GArgs s{};
      s.A = q16; s.aS1 = (long)Tt * Cc; s.aS2 = Dd; s.sa = Cc;
      s.Bm = k16; s.bS1 = (long)Tt * Cc; s.bS2 = Dd; s.sb = Cc;
      s.O = S; s.oS1 = (long)Hh * Tt * Tt; s.oS2 = (long)Tt * Tt; s.so = Tt;
      s.mask = mask; s.mS1 = (long)Tt * Tt; s.mS2 = 0;
      s.M = Tt; s.N = Tt; s.K = Dd; s.ZH = Hh; s.NZ = Bb * Hh;
      s.scale = 0.125f;  // d^-0.5
      rg<3, false>(s, stream);
    }
    softmax_k<<<Bb * Hh * Tt, 256, 0, stream>>>(S);
    transv_k<<<(MT * Cc) / 256, 256, 0, stream>>>(v16p, Vt);
    {  // O(b,h) = P V  -> o16 laid out as (B*T, C) with col = h*64+d
      GArgs s{};
      s.A = S; s.aS1 = (long)Hh * Tt * Tt; s.aS2 = (long)Tt * Tt; s.sa = Tt;
      s.Bm = Vt; s.bS1 = (long)Hh * Dd * Tt; s.bS2 = (long)Dd * Tt; s.sb = Tt;
      s.O = o16; s.oS1 = (long)Tt * Cc; s.oS2 = Dd; s.so = Cc;
      s.M = Tt; s.N = Dd; s.K = Tt; s.ZH = Hh; s.NZ = Bb * Hh;
      rg<4, false>(s, stream);
    }
    {  // x = x + o16 @ Wo + bo
      GArgs s{};
      s.A = o16; s.sa = Cc; s.Bm = woT[l]; s.sb = Cc;
      s.O = x; s.so = Cc; s.bias = LP(l, 9); s.res = x;
      s.M = MT; s.N = Cc; s.K = Cc; s.ZH = 1; s.NZ = 1;
      rg<2, false>(s, stream);
    }

    ln_k<<<MT, 256, 0, stream>>>(x, LP(l, 10), LP(l, 11), y16);
    {  // hffn = gelu(y @ W1 + b1)
      GArgs s{};
      s.A = y16; s.sa = Cc; s.Bm = w1T[l]; s.sb = Cc;
      s.O = hffn; s.so = FFNn; s.bias = LP(l, 13);
      s.M = MT; s.N = FFNn; s.K = Cc; s.ZH = 1; s.NZ = 1;
      rg<1, false>(s, stream);
    }
    {  // x = x + hffn @ W2 + b2
      GArgs s{};
      s.A = hffn; s.sa = FFNn; s.Bm = w2T[l]; s.sb = FFNn;
      s.O = x; s.so = Cc; s.bias = LP(l, 15); s.res = x;
      s.M = MT; s.N = Cc; s.K = FFNn; s.ZH = 1; s.NZ = 1;
      rg<2, false>(s, stream);
    }
  }

  // ---- final: out[b,:] = x[b, t=0, :] @ Wout + bout  (M=64 rows, fp32 A path) ----
  {
    GArgs s{};
    s.A = x; s.sa = Tt * Cc;  // row b -> x[b][0][:]
    s.Bm = woutT; s.sb = Cc;
    s.O = d_out; s.so = OUTC; s.bias = bout;
    s.M = Bb; s.N = OUTC; s.K = Cc; s.ZH = 1; s.NZ = 1;
    rg<5, true>(s, stream);
  }
}
